// MultiLabelContrastiveLoss_23673859736131
// MI455X (gfx1250) — compile-verified, matrix operand order checked by reading, not run on hardware
//
#include <hip/hip_runtime.h>
#include <hip/hip_bf16.h>
#include <math.h>

// ---------------------------------------------------------------------------
// MultiLabelContrastiveLoss on gfx1250 (MI455X)
//   loss = BCE(logits, labels) + contrastive(emb) with entropy weights.
//   sim = emb@emb.T (4096x4096x1024) via v_wmma_f32_16x16x32_f16, fused with
//   a branchless bounded-softmax epilogue (sim in [-14.3, 14.3], so exp(sim)
//   never over/underflows f32; row max only needed for the +1e-8 term).
// ---------------------------------------------------------------------------

typedef __attribute__((ext_vector_type(16))) _Float16 v16h;
typedef __attribute__((ext_vector_type(8)))  _Float16 v8h;
typedef __attribute__((ext_vector_type(8)))  float    v8f;

#define NROWS 4096
#define NDIM  1024
#define NCLS  64
#define NTILE (NROWS / 16)   // 256 column tiles of 16
#define NWAVES 8
#define INV_T (1.0f / 0.07f)

// ---- ws layout (bytes) ----
#define WS_EMBH   0                       // 4096*1024*2 = 8 MB f16 embeddings
#define WS_MASK   (8u * 1024u * 1024u)    // 4096 * 8 B  = 32 KB label bitmasks
#define WS_W      (WS_MASK + 32u * 1024u) // 4096 * 4 B  = 16 KB entropy weights
#define WS_ACC    (WS_W + 16u * 1024u)    // 4 floats: bce_sum, loss_sum, n_valid, pad

#define SHUF16(lo, hi) __builtin_shufflevector(lo, hi, 0,1,2,3,4,5,6,7,8,9,10,11,12,13,14,15)

__global__ void mlcl_init_accum(float* accum) {
    if (threadIdx.x < 4) accum[threadIdx.x] = 0.0f;
}

// f32 -> f16 conversion, 8 elements / thread (two b128 loads, one b128 store)
__global__ void mlcl_convert_f16(const float* __restrict__ src,
                                 _Float16* __restrict__ dst) {
    int t = blockIdx.x * blockDim.x + threadIdx.x;   // 0 .. N*D/8-1
    const float4* s = (const float4*)src + (size_t)t * 2;
    float4 x0 = s[0], x1 = s[1];
    v8h h;
    h[0] = (_Float16)x0.x; h[1] = (_Float16)x0.y;
    h[2] = (_Float16)x0.z; h[3] = (_Float16)x0.w;
    h[4] = (_Float16)x1.x; h[5] = (_Float16)x1.y;
    h[6] = (_Float16)x1.z; h[7] = (_Float16)x1.w;
    *(v8h*)(dst + (size_t)t * 8) = h;
}

// Per-row: BCE partial sum, entropy weight w[i], packed label bitmask.
__global__ void mlcl_row_prep(const float* __restrict__ logits,
                              const int* __restrict__ labels,
                              unsigned long long* __restrict__ masks,
                              float* __restrict__ w,
                              float* __restrict__ accum) {
    int i = blockIdx.x * blockDim.x + threadIdx.x;
    if (i >= NROWS) return;
    float bce = 0.0f, entsum = 0.0f, ysum = 0.0f;
    unsigned long long m = 0ull;
    #pragma unroll 4
    for (int c = 0; c < NCLS; ++c) {
        float l = logits[i * NCLS + c];
        int   y = labels[i * NCLS + c];
        float yf = (float)y;
        bce += fmaxf(l, 0.0f) - l * yf + log1pf(expf(-fabsf(l)));
        float p = 1.0f / (1.0f + expf(-l));
        float ent = -(p * logf(p + 1e-8f) + (1.0f - p) * logf(1.0f - p + 1e-8f));
        entsum += ent * yf;
        ysum += yf;
        m |= ((unsigned long long)(y != 0)) << c;
    }
    masks[i] = m;
    w[i] = entsum / (ysum + 1e-8f);
    atomicAdd(&accum[0], bce);
}

// Fused sim-GEMM + bounded-softmax contrastive loss.
// One block = one 16-row stripe; 8 wave32s each own PAIRS of column tiles:
// shared A fragment, two independent WMMA accumulator chains, register
// double-buffering so loads for step k+1 overlap the WMMAs of step k.
__global__ __launch_bounds__(256) void mlcl_contrastive(
    const _Float16* __restrict__ E,            // [N, D] f16, row-major
    const unsigned long long* __restrict__ masks,
    const float* __restrict__ w,
    float* __restrict__ accum) {
    __shared__ __align__(16) _Float16 ldsA[16 * NDIM];   // 32 KB A-stripe
    __shared__ float wstats[NWAVES][16][3];              // per-wave (max, neg, pos)

    const int ib    = blockIdx.x;          // row-tile index
    const int tid   = threadIdx.x;
    const int wave  = tid >> 5;
    const int lane  = tid & 31;
    const int n     = lane & 15;           // A-row / B-column / D-column per lane
    const int half  = lane >> 4;           // khalf select per ISA 16-bit layout
    const int mrow0 = half * 8;            // C/D: this half-wave holds rows mrow0..+7

    // stage 16 x 1024 f16 stripe into LDS (contiguous 32 KB copy)
    {
        const uint4* src = (const uint4*)(E + (size_t)ib * 16 * NDIM);
        uint4* dst = (uint4*)ldsA;
        for (int k = tid; k < 16 * NDIM / 8; k += 256) dst[k] = src[k];
    }
    __syncthreads();

    // hoist this lane's 8 row masks / row indices into registers (no LDS in hot loop)
    unsigned long long mi[8];
    int iRow[8];
    #pragma unroll
    for (int r = 0; r < 8; ++r) {
        iRow[r] = ib * 16 + mrow0 + r;
        mi[r]   = masks[iRow[r]];
    }

    float mrun[8], sneg[8], spos[8];
    #pragma unroll
    for (int r = 0; r < 8; ++r) { mrun[r] = -1e30f; sneg[r] = 0.0f; spos[r] = 0.0f; }

    const _Float16* arow = ldsA + (size_t)n * NDIM + half * 8;

    for (int jp = wave; jp < NTILE / 2; jp += NWAVES) {
        const int j0 = jp * 2, j1 = jp * 2 + 1;

        // prefetch next tile pair's B rows for this lane (global_prefetch path)
        const int jnp = jp + NWAVES;
        if (jnp < NTILE / 2) {
            __builtin_prefetch(E + (size_t)(jnp * 32 + n) * NDIM, 0, 3);
            __builtin_prefetch(E + (size_t)(jnp * 32 + 16 + n) * NDIM, 0, 3);
        }

        const unsigned long long mj0 = masks[j0 * 16 + n];
        const unsigned long long mj1 = masks[j1 * 16 + n];
        const _Float16* b0 = E + (size_t)(j0 * 16 + n) * NDIM + half * 8;
        const _Float16* b1 = E + (size_t)(j1 * 16 + n) * NDIM + half * 8;

        v8f acc0 = {}, acc1 = {};

        // register double-buffered K loop: 2 WMMAs per step, loads run ahead
        v8h aL[2], aH[2], b0L[2], b0H[2], b1L[2], b1H[2];
        aL[0]  = *(const v8h*)(arow);      aH[0]  = *(const v8h*)(arow + 16);
        b0L[0] = *(const v8h*)(b0);        b0H[0] = *(const v8h*)(b0 + 16);
        b1L[0] = *(const v8h*)(b1);        b1H[0] = *(const v8h*)(b1 + 16);

        #pragma unroll 4
        for (int kk = 0; kk < NDIM / 32; ++kk) {
            const int cur = kk & 1, nxt = cur ^ 1;
            if (kk + 1 < NDIM / 32) {
                const int o = (kk + 1) * 32;
                aL[nxt]  = *(const v8h*)(arow + o);  aH[nxt]  = *(const v8h*)(arow + o + 16);
                b0L[nxt] = *(const v8h*)(b0 + o);    b0H[nxt] = *(const v8h*)(b0 + o + 16);
                b1L[nxt] = *(const v8h*)(b1 + o);    b1H[nxt] = *(const v8h*)(b1 + o + 16);
            }
            v16h a  = SHUF16(aL[cur],  aH[cur]);
            v16h vb0 = SHUF16(b0L[cur], b0H[cur]);
            v16h vb1 = SHUF16(b1L[cur], b1H[cur]);
            acc0 = __builtin_amdgcn_wmma_f32_16x16x32_f16(
                false, a, false, vb0, (short)0, acc0, false, false);
            acc1 = __builtin_amdgcn_wmma_f32_16x16x32_f16(
                false, a, false, vb1, (short)0, acc1, false, false);
        }

        // branchless epilogue: sim bounded => plain exp sums + running max
        const int j0e = j0 * 16 + n, j1e = j1 * 16 + n;
        #pragma unroll
        for (int r = 0; r < 8; ++r) {
            float v0 = (iRow[r] != j0e) ? acc0[r] * INV_T : -1e30f;  // diag -> exp==0
            float v1 = (iRow[r] != j1e) ? acc1[r] * INV_T : -1e30f;
            float e0 = __expf(v0);
            float e1 = __expf(v1);
            sneg[r] += e0 + e1;
            spos[r] += (((mi[r] & mj0) != 0ull) ? e0 : 0.0f)
                     + (((mi[r] & mj1) != 0ull) ? e1 : 0.0f);
            mrun[r] = fmaxf(mrun[r], fmaxf(v0, v1));
        }
    }

    // reduce across the 16 lanes of each half-wave (plain sum / max)
    #pragma unroll
    for (int r = 0; r < 8; ++r) {
        for (int o = 8; o > 0; o >>= 1) {
            sneg[r] += __shfl_xor(sneg[r], o, 16);
            spos[r] += __shfl_xor(spos[r], o, 16);
            mrun[r] = fmaxf(mrun[r], __shfl_xor(mrun[r], o, 16));
        }
    }
    if (n == 0) {
        #pragma unroll
        for (int r = 0; r < 8; ++r) {
            wstats[wave][mrow0 + r][0] = mrun[r];
            wstats[wave][mrow0 + r][1] = sneg[r];
            wstats[wave][mrow0 + r][2] = spos[r];
        }
    }
    __syncthreads();

    // final per-row merge + loss accumulation (threads 0..15)
    if (tid < 16) {
        float M = -1e30f, SN = 0.0f, SP = 0.0f;
        #pragma unroll
        for (int wv = 0; wv < NWAVES; ++wv) {
            M = fmaxf(M, wstats[wv][tid][0]);
            SN += wstats[wv][tid][1];
            SP += wstats[wv][tid][2];
        }
        const int i = ib * 16 + tid;
        // ref: ratio = sum_pos e^{s-c} / (sum_neg e^{s-c} + 1e-8)
        //   == SP / (SN + 1e-8 * e^{c})   with our unshifted sums
        float ratio = SP / (SN + 1e-8f * __expf(M));
        float li    = -logf(ratio) * w[i];
        if (SP > 0.0f) {                        // pos_mask.any(axis=1)
            atomicAdd(&accum[1], li);
            atomicAdd(&accum[2], 1.0f);
        }
    }
}

__global__ void mlcl_finalize(const float* __restrict__ accum,
                              float* __restrict__ out) {
    float bce = accum[0] / (float)(NROWS * NCLS);
    float nv  = accum[2];
    float contrastive = (nv > 0.0f) ? (accum[1] / nv) : 0.0f;
    out[0] = bce + contrastive;                 // ALPHA = 1.0
}

extern "C" void kernel_launch(void* const* d_in, const int* in_sizes, int n_in,
                              void* d_out, int out_size, void* d_ws, size_t ws_size,
                              hipStream_t stream) {
    const float* emb    = (const float*)d_in[0];
    const float* logits = (const float*)d_in[1];
    const int*   labels = (const int*)d_in[2];
    float* out = (float*)d_out;

    char* ws = (char*)d_ws;
    _Float16*           embh  = (_Float16*)(ws + WS_EMBH);
    unsigned long long* masks = (unsigned long long*)(ws + WS_MASK);
    float*              wbuf  = (float*)(ws + WS_W);
    float*              accum = (float*)(ws + WS_ACC);

    mlcl_init_accum<<<1, 32, 0, stream>>>(accum);
    mlcl_convert_f16<<<(NROWS * NDIM / 8) / 256, 256, 0, stream>>>(emb, embh);
    mlcl_row_prep<<<NROWS / 256, 256, 0, stream>>>(logits, labels, masks, wbuf, accum);
    mlcl_contrastive<<<NTILE, 256, 0, stream>>>(embh, masks, wbuf, accum);
    mlcl_finalize<<<1, 1, 0, stream>>>(accum, out);
}
